// MBartMoEDecoderLayer_12446815223985
// MI455X (gfx1250) — matrix operand order repaired and checked by reading, not compiled, not run on hardware
//
#include <hip/hip_runtime.h>
#include <hip/hip_bf16.h>

// ---------------------------------------------------------------------------
// MBart MoE decoder layer for MI455X (gfx1250, wave32, WMMA + TDM).
// GEMMs: LDS-staged, double-buffered via tensor_load_to_lds (TENSORcnt),
// compute via v_wmma_f32_16x16x32_bf16.
// ---------------------------------------------------------------------------

typedef __bf16 bf16_t;
typedef __attribute__((ext_vector_type(16))) __bf16 v16bf;
typedef __attribute__((ext_vector_type(8)))  __bf16 v8bf;
typedef __attribute__((ext_vector_type(8)))  float  v8f;
typedef __attribute__((ext_vector_type(4)))  unsigned int v4u;
typedef __attribute__((ext_vector_type(8)))  int v8i;
typedef __attribute__((ext_vector_type(4)))  int v4i;

#define NEG_INF_SCORE (-1e9f)

#if defined(__has_builtin)
#if __has_builtin(__builtin_amdgcn_tensor_load_to_lds) && \
    __has_builtin(__builtin_amdgcn_s_wait_tensorcnt)
#define USE_TDM 1
#endif
#endif
#ifndef USE_TDM
#define USE_TDM 0
#endif

__device__ inline v8f wmma_bf16(v16bf a, v16bf b, v8f c) {
    return __builtin_amdgcn_wmma_f32_16x16x32_bf16(
        /*neg_a=*/false, a, /*neg_b=*/false, b,
        /*c_mod=*/(short)0, c, /*reuse_a=*/false, /*reuse_b=*/false);
}

// A fragment, 16x32 bf16, row-major source.
// ISA layout: lanes 0-15 row m hold K in [h*8,h*8+8) U [16+h*8,16+h*8+8).
__device__ inline v16bf load_a_frag(const bf16_t* __restrict__ base, int ld,
                                    size_t row0, int k0) {
    int lane = threadIdx.x & 31;
    int m = lane & 15, hh = lane >> 4;
    const bf16_t* p = base + (row0 + (size_t)m) * (size_t)ld + k0 + hh * 8;
    v8bf c0 = *(const v8bf*)(p);
    v8bf c1 = *(const v8bf*)(p + 16);
    return __builtin_shufflevector(c0, c1, 0, 1, 2, 3, 4, 5, 6, 7,
                                   8, 9, 10, 11, 12, 13, 14, 15);
}

// B fragment, 32x16 bf16, from K-contiguous rows ("B^T" rows of length ld):
// lane n = column, lanes hold K in [h*8,h*8+8) U [16+h*8,16+h*8+8).
__device__ inline v16bf load_b_frag(const bf16_t* __restrict__ base, int ld,
                                    size_t n0, int k0) {
    int lane = threadIdx.x & 31;
    int n = lane & 15, hh = lane >> 4;
    const bf16_t* p = base + (n0 + (size_t)n) * (size_t)ld + k0 + hh * 8;
    v8bf c0 = *(const v8bf*)(p);
    v8bf c1 = *(const v8bf*)(p + 16);
    return __builtin_shufflevector(c0, c1, 0, 1, 2, 3, 4, 5, 6, 7,
                                   8, 9, 10, 11, 12, 13, 14, 15);
}

#if USE_TDM
// Issue a 2-D TDM tile load: tile_d1 rows x 32 bf16 from a row-major tensor
// with row stride `strideElems` into packed LDS [rows][32].
// D# layout per CDNA5 ISA ch.8 (group0: count/lds/global/type, group1: dims).
__device__ inline void tdm_load_tile_2d(const bf16_t* gsrc,
                                        unsigned int lds_addr,
                                        int tile_d1, long long strideElems) {
    unsigned long long ga = (unsigned long long)(size_t)gsrc;
    v4u g0 = {
        1u,                                           // count=1, user desc
        lds_addr,                                     // lds_addr[31:0]
        (unsigned int)(ga & 0xFFFFFFFFull),           // global_addr[31:0]
        (unsigned int)((ga >> 32) & 0x1FFFFFFull)     // global_addr[56:32]
            | (2u << 30)                              // type=2 (image)
    };
    int d0 = 0x40000000;                              // tensor_dim0 (huge)
    int d1 = 0x40000000;                              // tensor_dim1 (huge)
    v8i g1 = {
        (int)(1u << 16),                              // data_size=1 (2B), mask=0
        (int)((d0 & 0xFFFF) << 16),                   // dim0[15:0]
        (int)(((unsigned)d0 >> 16) | ((d1 & 0xFFFF) << 16)),
        (int)(((unsigned)d1 >> 16) | (32u << 16)),    // tile_dim0 = 32
        (int)(tile_d1 & 0xFFFF),                      // tile_dim1, tile_dim2=0
        (int)(strideElems & 0xFFFFFFFFll),            // dim0_stride[31:0]
        (int)((strideElems >> 32) & 0xFFFFll),        // dim0_stride[47:32]
        0
    };
    v4i z4 = {0, 0, 0, 0};
    v8i z8 = {0, 0, 0, 0, 0, 0, 0, 0};
    __builtin_amdgcn_tensor_load_to_lds(g0, g1, z4, z4, z8, 0);
}
#endif

// ---------------------------------------------------------------------------
// LDS-staged WMMA GEMM: out = epilogue((A[M,K]bf16 @ B[K,N]bf16 + bias)*alpha)
// B supplied transposed: Bt[N,K].  Block = 256 thr (8 waves), tile 64x128.
// Per K-step, the 64x32 A tile and 128x32 B tile are staged in LDS
// (double-buffered TDM on TENSORcnt when available, cooperative copy else),
// then each wave runs 4 WMMAs on its 16x64 strip.
// mode 0: bf16   mode 1: f32 (+residual)   mode 2: bf16 transposed per batch
// mode 3: f32 += gate[(row/rpb)*8+expert] * v
// ---------------------------------------------------------------------------
__global__ void gemm_wmma(const bf16_t* __restrict__ A,
                          const bf16_t* __restrict__ Bt,
                          const float* __restrict__ bias,
                          const float* __restrict__ residual,
                          const float* __restrict__ gate,
                          void* __restrict__ out,
                          int N, int K, float alpha, int mode, int expert,
                          int rpb) {
    int wave = threadIdx.x >> 5;
    int lane = threadIdx.x & 31;
    int half = lane >> 4, nn = lane & 15;
    int rowB = blockIdx.x * 64;
    int colB = blockIdx.y * 128;
    int row0 = rowB + (wave & 3) * 16;
    int col0 = colB + (wave >> 2) * 64;

    v8f acc[4];
#pragma unroll
    for (int j = 0; j < 4; ++j)
#pragma unroll
        for (int i = 0; i < 8; ++i) acc[j][i] = 0.f;

#if USE_TDM
    __shared__ __align__(16) bf16_t shA[2][64 * 32];    // 8 KB
    __shared__ __align__(16) bf16_t shB[2][128 * 32];   // 16 KB
    if (wave == 0) {  // prime buffer 0
        tdm_load_tile_2d(A + (size_t)rowB * K, (unsigned int)(size_t)&shA[0][0],
                         64, K);
        tdm_load_tile_2d(Bt + (size_t)colB * K, (unsigned int)(size_t)&shB[0][0],
                         128, K);
    }
    int it = 0;
    for (int k0 = 0; k0 < K; k0 += 32, ++it) {
        int cur = it & 1;
        bool hasNext = (k0 + 32) < K;
        if (wave == 0) {
            if (hasNext) {  // prefetch next K-chunk into the other buffer
                tdm_load_tile_2d(A + (size_t)rowB * K + k0 + 32,
                                 (unsigned int)(size_t)&shA[cur ^ 1][0], 64, K);
                tdm_load_tile_2d(Bt + (size_t)colB * K + k0 + 32,
                                 (unsigned int)(size_t)&shB[cur ^ 1][0], 128, K);
                // 2 prefetch TDMs may stay in flight; current 2 must be done.
                __builtin_amdgcn_s_wait_tensorcnt(2);
            } else {
                __builtin_amdgcn_s_wait_tensorcnt(0);
            }
        }
        __syncthreads();  // publish tile to all waves
        v16bf a = load_a_frag(&shA[cur][0], 32, (size_t)((wave & 3) * 16), 0);
#pragma unroll
        for (int j = 0; j < 4; ++j) {
            v16bf b = load_b_frag(&shB[cur][0], 32,
                                  (size_t)((wave >> 2) * 64 + j * 16), 0);
            acc[j] = wmma_bf16(a, b, acc[j]);
        }
        __syncthreads();  // reads done before buffer is overwritten
    }
#else
    __shared__ __align__(16) bf16_t shA[64 * 32];
    __shared__ __align__(16) bf16_t shB[128 * 32];
    for (int k0 = 0; k0 < K; k0 += 32) {
        __syncthreads();
        {   // cooperative stage: A 64 rows x 4 chunks of 16B, B 128 rows x 4
            int t = threadIdx.x;
            int ar = t >> 2, ac = t & 3;
            *(v8bf*)(&shA[ar * 32 + ac * 8]) =
                *(const v8bf*)(A + (size_t)(rowB + ar) * K + k0 + ac * 8);
#pragma unroll
            for (int u = 0; u < 2; ++u) {
                int idx = t + u * 256;
                int br = idx >> 2, bc = idx & 3;
                *(v8bf*)(&shB[br * 32 + bc * 8]) =
                    *(const v8bf*)(Bt + (size_t)(colB + br) * K + k0 + bc * 8);
            }
        }
        __syncthreads();
        v16bf a = load_a_frag(shA, 32, (size_t)((wave & 3) * 16), 0);
#pragma unroll
        for (int j = 0; j < 4; ++j) {
            v16bf b = load_b_frag(shB, 32,
                                  (size_t)((wave >> 2) * 64 + j * 16), 0);
            acc[j] = wmma_bf16(a, b, acc[j]);
        }
    }
#endif

#pragma unroll
    for (int j = 0; j < 4; ++j) {
#pragma unroll
        for (int i = 0; i < 8; ++i) {
            int row = row0 + i + half * 8;
            int col = col0 + j * 16 + nn;
            float v = acc[j][i];
            if (bias) v += bias[col];
            v *= alpha;
            size_t idx = (size_t)row * N + col;
            if (mode == 0) {
                ((bf16_t*)out)[idx] = (bf16_t)v;
            } else if (mode == 1) {
                float r = residual ? residual[idx] : 0.f;
                ((float*)out)[idx] = v + r;
            } else if (mode == 2) {
                int bb = row / rpb, tt = row - bb * rpb;
                ((bf16_t*)out)[((size_t)bb * N + col) * rpb + tt] = (bf16_t)v;
            } else {
                float sc = gate[(row / rpb) * 8 + expert];
                ((float*)out)[idx] += sc * v;
            }
        }
    }
}

// ---------------------------------------------------------------------------
// Flash attention (GQA 16h/4kv, hd=64).  Block = 128 thr (4 waves), each wave
// owns a 16-query tile; 32 keys/step: 4 WMMA QK^T + online softmax (16-lane
// shfl reductions) + P relayout via LDS + 4 WMMA PV.
// Q [B*Sq,1024] (SCALING pre-applied), K [B*Sk,256], Vt [B,4,64,Sk].
// ---------------------------------------------------------------------------
__global__ void flash_attn_gqa(const bf16_t* __restrict__ Q,
                               const bf16_t* __restrict__ Kb,
                               const bf16_t* __restrict__ Vt,
                               bf16_t* __restrict__ O,
                               int Sq, int Sk, int causal) {
    constexpr int Hh = 16, KVH = 4, HD = 64;
    int bh = blockIdx.x;
    int b = bh / Hh, h = bh % Hh;
    int kvh = h / (Hh / KVH);
    int wave = threadIdx.x >> 5;
    int lane = threadIdx.x & 31;
    int half = lane >> 4, nn = lane & 15;
    int q0 = blockIdx.y * 64 + wave * 16;
    const int ldq = Hh * HD;
    const int ldk = KVH * HD;

    size_t qrow0 = (size_t)b * Sq + q0;
    v16bf qf0 = load_a_frag(Q, ldq, qrow0, h * HD + 0);
    v16bf qf1 = load_a_frag(Q, ldq, qrow0, h * HD + 32);

    float mrow[8], lrow[8];
    v8f oacc[4];
#pragma unroll
    for (int i = 0; i < 8; ++i) { mrow[i] = -3e30f; lrow[i] = 0.f; }
#pragma unroll
    for (int j = 0; j < 4; ++j)
#pragma unroll
        for (int i = 0; i < 8; ++i) oacc[j][i] = 0.f;

    __shared__ __align__(16) bf16_t psh[4][16][32];

    int blockEnd = (int)(blockIdx.y) * 64 + 64;
    int kend = causal ? (Sk < blockEnd ? Sk : blockEnd) : Sk;
    size_t krow0 = (size_t)b * Sk;
    const bf16_t* vbase = Vt + ((size_t)b * KVH + kvh) * HD * (size_t)Sk;

    for (int kb = 0; kb < kend; kb += 32) {
        v8f s0, s1;
#pragma unroll
        for (int i = 0; i < 8; ++i) { s0[i] = 0.f; s1[i] = 0.f; }
        s0 = wmma_bf16(qf0, load_b_frag(Kb, ldk, krow0 + kb, kvh * HD + 0), s0);
        s0 = wmma_bf16(qf1, load_b_frag(Kb, ldk, krow0 + kb, kvh * HD + 32), s0);
        s1 = wmma_bf16(qf0, load_b_frag(Kb, ldk, krow0 + kb + 16, kvh * HD + 0), s1);
        s1 = wmma_bf16(qf1, load_b_frag(Kb, ldk, krow0 + kb + 16, kvh * HD + 32), s1);

        if (causal) {
#pragma unroll
            for (int i = 0; i < 8; ++i) {
                int qg = q0 + i + half * 8;
                if (kb + nn > qg) s0[i] = NEG_INF_SCORE;
                if (kb + 16 + nn > qg) s1[i] = NEG_INF_SCORE;
            }
        }

#pragma unroll
        for (int i = 0; i < 8; ++i) {
            float rm = fmaxf(s0[i], s1[i]);
#pragma unroll
            for (int off = 8; off > 0; off >>= 1)
                rm = fmaxf(rm, __shfl_xor(rm, off, 16));
            float mn = fmaxf(mrow[i], rm);
            float sc = __expf(mrow[i] - mn);
            mrow[i] = mn;
            float p0 = __expf(s0[i] - mn);
            float p1 = __expf(s1[i] - mn);
            s0[i] = p0; s1[i] = p1;
            float rs = p0 + p1;
#pragma unroll
            for (int off = 8; off > 0; off >>= 1)
                rs += __shfl_xor(rs, off, 16);
            lrow[i] = lrow[i] * sc + rs;
#pragma unroll
            for (int j = 0; j < 4; ++j) oacc[j][i] *= sc;
        }

        __syncthreads();
#pragma unroll
        for (int i = 0; i < 8; ++i) {
            psh[wave][i + half * 8][nn]      = (bf16_t)s0[i];
            psh[wave][i + half * 8][nn + 16] = (bf16_t)s1[i];
        }
        __syncthreads();
        v16bf pf = load_a_frag(&psh[wave][0][0], 32, 0, 0);

#pragma unroll
        for (int j = 0; j < 4; ++j) {
            v16bf vf = load_b_frag(vbase, Sk, (size_t)(j * 16), kb);
            oacc[j] = wmma_bf16(pf, vf, oacc[j]);
        }
    }

#pragma unroll
    for (int j = 0; j < 4; ++j)
#pragma unroll
        for (int i = 0; i < 8; ++i) {
            size_t row = (size_t)b * Sq + q0 + i + half * 8;
            int col = h * HD + j * 16 + nn;
            O[row * ldq + col] = (bf16_t)(oacc[j][i] / lrow[i]);
        }
}

// ---------------------------------------------------------------------------
__global__ void layernorm_to_bf16(const float* __restrict__ x,
                                  const float* __restrict__ g,
                                  const float* __restrict__ bta,
                                  bf16_t* __restrict__ out, int D) {
    int row = blockIdx.x;
    const float* xr = x + (size_t)row * D;
    float s1 = 0.f, s2 = 0.f;
    for (int i = threadIdx.x; i < D; i += blockDim.x) {
        float v = xr[i];
        s1 += v; s2 += v * v;
    }
#pragma unroll
    for (int off = 16; off > 0; off >>= 1) {
        s1 += __shfl_xor(s1, off, 32);
        s2 += __shfl_xor(s2, off, 32);
    }
    __shared__ float sh1[8], sh2[8];
    int wave = threadIdx.x >> 5;
    if ((threadIdx.x & 31) == 0) { sh1[wave] = s1; sh2[wave] = s2; }
    __syncthreads();
    float t1 = 0.f, t2 = 0.f;
#pragma unroll
    for (int w = 0; w < 8; ++w) { t1 += sh1[w]; t2 += sh2[w]; }
    float mean = t1 / D;
    float var = t2 / D - mean * mean;
    float inv = rsqrtf(var + 1e-5f);
    bf16_t* orow = out + (size_t)row * D;
    for (int i = threadIdx.x; i < D; i += blockDim.x)
        orow[i] = (bf16_t)((xr[i] - mean) * inv * g[i] + bta[i]);
}

__global__ void cvt_f32_bf16(const float* __restrict__ in,
                             bf16_t* __restrict__ out, size_t n) {
    size_t idx = (size_t)blockIdx.x * blockDim.x + threadIdx.x;
    if (idx < n) out[idx] = (bf16_t)in[idx];
}

// batched transpose+convert: W[e][K][N] fp32 -> Wt[e][N][K] bf16
__global__ void cvt_transpose_w(const float* __restrict__ W,
                                bf16_t* __restrict__ Wt,
                                int K, int N, size_t n) {
    size_t idx = (size_t)blockIdx.x * blockDim.x + threadIdx.x;
    if (idx >= n) return;
    size_t KN = (size_t)K * N;
    size_t e = idx / KN;
    size_t r = idx - e * KN;
    int nc = (int)(r / K);
    int kc = (int)(r - (size_t)nc * K);
    Wt[e * KN + (size_t)nc * K + kc] = (bf16_t)W[e * KN + (size_t)kc * N + nc];
}

__global__ void gelu_mul_bf16(const float* __restrict__ t1,
                              const float* __restrict__ t2,
                              bf16_t* __restrict__ out, size_t n) {
    size_t idx = (size_t)blockIdx.x * blockDim.x + threadIdx.x;
    if (idx >= n) return;
    float x = t1[idx];
    float ge = 0.5f * x * (1.0f + erff(x * 0.70710678118654752f));
    out[idx] = (bf16_t)(ge * t2[idx]);
}

__global__ void gate_kernel(const int* __restrict__ langs,
                            const int* __restrict__ codes,
                            float* __restrict__ gate, int B, int E, int L) {
    int tid = threadIdx.x;
    if (tid >= B * E) return;
    int b = tid / E, e = tid % E;
    int cnt = 0; bool hit = false;
    for (int l = 0; l < L; ++l) {
        int lg = langs[b * L + l];
        cnt += (lg > 3) ? 1 : 0;
        hit = hit || (lg == codes[e]);
    }
    float rw = (cnt == 0) ? 1.f : 1.f / (float)cnt;
    gate[tid] = hit ? rw : 0.f;
}

// ---------------------------------------------------------------------------
extern "C" void kernel_launch(void* const* d_in, const int* in_sizes, int n_in,
                              void* d_out, int out_size, void* d_ws, size_t ws_size,
                              hipStream_t stream) {
    (void)in_sizes; (void)n_in; (void)out_size; (void)ws_size;
    constexpr int B = 4, S = 1024, SENC = 1024, D = 1024;
    constexpr int H = 16, HKV = 4, HD = 64, E = 8, F = 2048, L = 2;
    constexpr float SCALING = 0.125f;  // 64^-0.5
    const int M = B * S;
    const int MK = B * SENC;

    const float* hs   = (const float*)d_in[0];
    const float* enc  = (const float*)d_in[1];
    /* d_in[2] = attention_mask: exact causal mask, applied analytically */
    const int*   langs = (const int*)d_in[3];
    const int*   codes = (const int*)d_in[4];
    const float* ln1g = (const float*)d_in[5],  *ln1b = (const float*)d_in[6];
    const float* ln2g = (const float*)d_in[7],  *ln2b = (const float*)d_in[8];
    const float* ln3g = (const float*)d_in[9],  *ln3b = (const float*)d_in[10];
    const float* sa_qw = (const float*)d_in[11], *sa_qb = (const float*)d_in[12];
    const float* sa_kw = (const float*)d_in[13], *sa_kb = (const float*)d_in[14];
    const float* sa_vw = (const float*)d_in[15], *sa_vb = (const float*)d_in[16];
    const float* sa_ow = (const float*)d_in[17], *sa_ob = (const float*)d_in[18];
    const float* ca_qw = (const float*)d_in[19], *ca_qb = (const float*)d_in[20];
    const float* ca_kw = (const float*)d_in[21], *ca_kb = (const float*)d_in[22];
    const float* ca_vw = (const float*)d_in[23], *ca_vb = (const float*)d_in[24];
    const float* ca_ow = (const float*)d_in[25], *ca_ob = (const float*)d_in[26];
    const float* moe_w1 = (const float*)d_in[27];
    const float* moe_w2 = (const float*)d_in[28];
    const float* moe_w3 = (const float*)d_in[29];
    float* outp = (float*)d_out;

    size_t off = 0;
    auto carve = [&](size_t bytes) -> void* {
        void* p = (char*)d_ws + off;
        off += (bytes + 255) & ~(size_t)255;
        return p;
    };
    bf16_t* wt_saq = (bf16_t*)carve((size_t)D * D * 2);
    bf16_t* wt_sak = (bf16_t*)carve((size_t)D * HKV * HD * 2);
    bf16_t* wt_sav = (bf16_t*)carve((size_t)D * HKV * HD * 2);
    bf16_t* wt_sao = (bf16_t*)carve((size_t)D * D * 2);
    bf16_t* wt_caq = (bf16_t*)carve((size_t)D * D * 2);
    bf16_t* wt_cak = (bf16_t*)carve((size_t)D * HKV * HD * 2);
    bf16_t* wt_cav = (bf16_t*)carve((size_t)D * HKV * HD * 2);
    bf16_t* wt_cao = (bf16_t*)carve((size_t)D * D * 2);
    bf16_t* wt_m1 = (bf16_t*)carve((size_t)E * D * F * 2);
    bf16_t* wt_m2 = (bf16_t*)carve((size_t)E * F * D * 2);
    bf16_t* wt_m3 = (bf16_t*)carve((size_t)E * D * F * 2);
    bf16_t* enc_bf  = (bf16_t*)carve((size_t)MK * D * 2);
    bf16_t* ln_bf   = (bf16_t*)carve((size_t)M * D * 2);
    bf16_t* q_bf    = (bf16_t*)carve((size_t)M * D * 2);
    bf16_t* k_bf    = (bf16_t*)carve((size_t)MK * HKV * HD * 2);
    bf16_t* vt_bf   = (bf16_t*)carve((size_t)MK * HKV * HD * 2);
    bf16_t* attn_bf = (bf16_t*)carve((size_t)M * D * 2);
    float*  h1 = (float*)carve((size_t)M * D * 4);
    float*  h2 = (float*)carve((size_t)M * D * 4);
    float*  t1 = (float*)carve((size_t)M * F * 4);
    float*  t2 = (float*)carve((size_t)M * F * 4);
    bf16_t* hff = (bf16_t*)carve((size_t)M * F * 2);
    float*  gate = (float*)carve((size_t)B * E * 4);

    auto xpose = [&](const float* W, bf16_t* Wt, int K, int N, int Eb) {
        size_t n = (size_t)Eb * K * N;
        cvt_transpose_w<<<(unsigned)((n + 255) / 256), 256, 0, stream>>>(W, Wt, K, N, n);
    };
    auto gemm = [&](const bf16_t* A, const bf16_t* Bt, const float* bias,
                    const float* res, const float* gp, void* out,
                    int Mi, int Ni, int Ki, float alpha, int mode, int e, int rpb) {
        dim3 grid(Mi / 64, Ni / 128);
        gemm_wmma<<<grid, 256, 0, stream>>>(A, Bt, bias, res, gp, out,
                                            Ni, Ki, alpha, mode, e, rpb);
    };

    // -- weight / input conversions ------------------------------------------
    xpose(sa_qw, wt_saq, D, D, 1);
    xpose(sa_kw, wt_sak, D, HKV * HD, 1);
    xpose(sa_vw, wt_sav, D, HKV * HD, 1);
    xpose(sa_ow, wt_sao, D, D, 1);
    xpose(ca_qw, wt_caq, D, D, 1);
    xpose(ca_kw, wt_cak, D, HKV * HD, 1);
    xpose(ca_vw, wt_cav, D, HKV * HD, 1);
    xpose(ca_ow, wt_cao, D, D, 1);
    xpose(moe_w1, wt_m1, D, F, E);
    xpose(moe_w2, wt_m2, F, D, E);
    xpose(moe_w3, wt_m3, D, F, E);
    {
        size_t n = (size_t)MK * D;
        cvt_f32_bf16<<<(unsigned)((n + 255) / 256), 256, 0, stream>>>(enc, enc_bf, n);
    }
    gate_kernel<<<1, 32, 0, stream>>>(langs, codes, gate, B, E, L);

    // -- self attention ------------------------------------------------------
    layernorm_to_bf16<<<M, 256, 0, stream>>>(hs, ln1g, ln1b, ln_bf, D);
    gemm(ln_bf, wt_saq, sa_qb, nullptr, nullptr, q_bf, M, D, D, SCALING, 0, 0, 0);
    gemm(ln_bf, wt_sak, sa_kb, nullptr, nullptr, k_bf, M, HKV * HD, D, 1.f, 0, 0, 0);
    gemm(ln_bf, wt_sav, sa_vb, nullptr, nullptr, vt_bf, M, HKV * HD, D, 1.f, 2, 0, S);
    flash_attn_gqa<<<dim3(B * H, S / 64), 128, 0, stream>>>(q_bf, k_bf, vt_bf,
                                                            attn_bf, S, S, 1);
    gemm(attn_bf, wt_sao, sa_ob, hs, nullptr, h1, M, D, D, 1.f, 1, 0, 0);

    // -- cross attention -----------------------------------------------------
    layernorm_to_bf16<<<M, 256, 0, stream>>>(h1, ln2g, ln2b, ln_bf, D);
    gemm(ln_bf, wt_caq, ca_qb, nullptr, nullptr, q_bf, M, D, D, SCALING, 0, 0, 0);
    gemm(enc_bf, wt_cak, ca_kb, nullptr, nullptr, k_bf, MK, HKV * HD, D, 1.f, 0, 0, 0);
    gemm(enc_bf, wt_cav, ca_vb, nullptr, nullptr, vt_bf, MK, HKV * HD, D, 1.f, 2, 0, SENC);
    flash_attn_gqa<<<dim3(B * H, S / 64), 128, 0, stream>>>(q_bf, k_bf, vt_bf,
                                                            attn_bf, S, SENC, 0);
    gemm(attn_bf, wt_cao, ca_ob, h1, nullptr, h2, M, D, D, 1.f, 1, 0, 0);

    // -- MoE FFN (all experts dense, gated accumulate) -----------------------
    (void)hipMemcpyAsync(outp, h2, (size_t)M * D * sizeof(float),
                         hipMemcpyDeviceToDevice, stream);  // out = residual
    layernorm_to_bf16<<<M, 256, 0, stream>>>(h2, ln3g, ln3b, ln_bf, D);
    for (int e = 0; e < E; ++e) {
        const bf16_t* w1e = wt_m1 + (size_t)e * D * F;  // [F, D]
        const bf16_t* w3e = wt_m3 + (size_t)e * D * F;  // [F, D]
        const bf16_t* w2e = wt_m2 + (size_t)e * F * D;  // [D, F]
        gemm(ln_bf, w1e, nullptr, nullptr, nullptr, t1, M, F, D, 1.f, 1, 0, 0);
        gemm(ln_bf, w3e, nullptr, nullptr, nullptr, t2, M, F, D, 1.f, 1, 0, 0);
        {
            size_t n = (size_t)M * F;
            gelu_mul_bf16<<<(unsigned)((n + 255) / 256), 256, 0, stream>>>(t1, t2, hff, n);
        }
        gemm(hff, w2e, nullptr, nullptr, gate, outp, M, D, F, 1.f, 3, e, S);
    }
}
